// SASREC_41790031790819
// MI455X (gfx1250) — matrix operand order repaired and checked by reading, not compiled, not run on hardware
//
#include <hip/hip_runtime.h>
#include <hip/hip_bf16.h>
#include <cstdint>

// ---------------- problem constants ----------------
namespace {
constexpr int B_ = 64, S_ = 512, E_ = 512, H_ = 8, D_ = 64, F_ = 512, L_ = 2;
constexpr int BS = B_ * S_;                  // 32768 rows
constexpr float SQRT_E = 22.627416997969522f;
constexpr float PADV   = -4294967296.0f;     // float(-2^32+1)
}

typedef __bf16 bf16_t;
typedef __attribute__((ext_vector_type(16))) __bf16 v16bf;
typedef __attribute__((ext_vector_type(8)))  __bf16 v8bf;
typedef __attribute__((ext_vector_type(8)))  float  v8f;
typedef __attribute__((ext_vector_type(4)))  unsigned int ui32x4;
typedef __attribute__((ext_vector_type(8)))  int i32x8;
typedef __attribute__((ext_vector_type(4)))  int i32x4;

// ---------------- WMMA helpers ----------------
static __device__ inline v16bf join8(v8bf lo, v8bf hi) {
  return __builtin_shufflevector(lo, hi, 0,1,2,3,4,5,6,7,8,9,10,11,12,13,14,15);
}

static __device__ inline v8f wmma_bf16(v16bf a, v16bf b, v8f c) {
  // D = A(16x32) * B(32x16) + C(16x16) , f32 accumulate
  return __builtin_amdgcn_wmma_f32_16x16x32_bf16(false, a, false, b, (short)0, c,
                                                 false, false);
}

// A fragment: 16x32 bf16 tile, row-major source with leading dim `ld`.
// Contiguous 16-bit A layout: lane -> row = lane&15, kbase = (lane>=16)?16:0,
// elems 0..15 -> K=kbase+e.
static __device__ inline v16bf load_a_frag(const bf16_t* __restrict__ base, int ld) {
  int lane = threadIdx.x & 31;
  const bf16_t* p = base + (size_t)(lane & 15) * ld + ((lane >> 4) ? 16 : 0);
  v8bf lo = *(const v8bf*)(p);
  v8bf hi = *(const v8bf*)(p + 8);
  return join8(lo, hi);
}

// B fragment for C = A * Bsrc^T where Bsrc is row-major (N rows x K cols):
// lane -> n = lane&15, K = ((lane>=16)?16:0) + e  (16 contiguous elements).
static __device__ inline v16bf load_bt_frag(const bf16_t* __restrict__ base, int ld) {
  int lane = threadIdx.x & 31;
  const bf16_t* p = base + (size_t)(lane & 15) * ld + ((lane >> 4) ? 16 : 0);
  v8bf lo = *(const v8bf*)(p);
  v8bf hi = *(const v8bf*)(p + 8);
  return join8(lo, hi);
}

// B fragment from a pre-packed tile: 32 contiguous bytes per lane.
static __device__ inline v16bf load_packed_b(const bf16_t* tile) {
  int lane = threadIdx.x & 31;
  const bf16_t* p = tile + lane * 16;
  v8bf lo = *(const v8bf*)(p);
  v8bf hi = *(const v8bf*)(p + 8);
  return join8(lo, hi);
}

// ---------------- weight packing ----------------
// Pack W (512x512 f32, k-major) into bf16 tiles (kt,nt) of 32x16 such that each
// lane's 16 B-fragment elements are contiguous:
//   packed[((kt*32+nt)*32 + lane)*16 + e] = W[(kt*32 + 16*(lane>=16) + e)*512 + nt*16 + (lane&15)]
__global__ void __launch_bounds__(256)
pack_w_kernel(const float* __restrict__ W, bf16_t* __restrict__ P) {
  int t = blockIdx.x * 256 + threadIdx.x;           // 262144 total
  int e    = t & 15;
  int lane = (t >> 4) & 31;
  int nt   = (t >> 9) & 31;
  int kt   = t >> 14;
  int k = kt * 32 + ((lane >= 16) ? 16 : 0) + e;
  int n = nt * 16 + (lane & 15);
  P[t] = (bf16_t)W[(size_t)k * 512 + n];
}

// ---------------- embedding ----------------
__global__ void __launch_bounds__(256)
embed_kernel(const int* __restrict__ seq, const float* __restrict__ item_emb,
             const float* __restrict__ pos_emb, float* __restrict__ x) {
  size_t t = (size_t)blockIdx.x * 256 + threadIdx.x;   // BS*E threads
  int e   = (int)(t & 511);
  int row = (int)(t >> 9);
  int s   = row & (S_ - 1);
  int id  = seq[row];
  float m = (id != 0) ? 1.f : 0.f;
  x[t] = (item_emb[(size_t)id * E_ + e] * SQRT_E + pos_emb[(size_t)s * E_ + e]) * m;
}

// ---------------- f32 -> bf16 row convert (+ optional key mask) ----------------
__global__ void __launch_bounds__(256)
rowconv_kernel(const float* __restrict__ x, bf16_t* __restrict__ xb,
               float* __restrict__ km) {
  __shared__ float red[256];
  int row = blockIdx.x, tid = threadIdx.x;
  size_t base = (size_t)row * E_;
  float v0 = x[base + tid], v1 = x[base + 256 + tid];
  xb[base + tid]       = (bf16_t)v0;
  xb[base + 256 + tid] = (bf16_t)v1;
  if (km != nullptr) {
    red[tid] = v0 + v1;
    __syncthreads();
    for (int i = 128; i > 0; i >>= 1) {
      if (tid < i) red[tid] += red[tid + i];
      __syncthreads();
    }
    if (tid == 0) km[row] = (red[0] != 0.f) ? 1.f : 0.f;
  }
}

// ---------------- layernorm (per-position gamma/beta) ----------------
__global__ void __launch_bounds__(256)
ln_kernel(const float* __restrict__ x, const float* __restrict__ gamma,
          const float* __restrict__ beta, float* __restrict__ xn,
          bf16_t* __restrict__ xnb, float* __restrict__ qm) {
  __shared__ float r1[256], r2[256];
  int row = blockIdx.x, tid = threadIdx.x;
  int s = row & (S_ - 1);
  size_t base = (size_t)row * E_;
  float v0 = x[base + tid], v1 = x[base + 256 + tid];
  r1[tid] = v0 + v1;
  r2[tid] = v0 * v0 + v1 * v1;
  __syncthreads();
  for (int i = 128; i > 0; i >>= 1) {
    if (tid < i) { r1[tid] += r1[tid + i]; r2[tid] += r2[tid + i]; }
    __syncthreads();
  }
  float mean = r1[0] * (1.f / E_);
  float var  = r2[0] * (1.f / E_) - mean * mean;
  float rstd = rsqrtf(var + 1e-8f);
  __syncthreads();
  size_t gb = (size_t)s * E_;
  float y0 = gamma[gb + tid]       * ((v0 - mean) * rstd) + beta[gb + tid];
  float y1 = gamma[gb + 256 + tid] * ((v1 - mean) * rstd) + beta[gb + 256 + tid];
  xn[base + tid]       = y0;
  xn[base + 256 + tid] = y1;
  if (xnb != nullptr) {
    xnb[base + tid]       = (bf16_t)y0;
    xnb[base + 256 + tid] = (bf16_t)y1;
  }
  if (qm != nullptr) {
    r1[tid] = y0 + y1;
    __syncthreads();
    for (int i = 128; i > 0; i >>= 1) {
      if (tid < i) r1[tid] += r1[tid + i];
      __syncthreads();
    }
    if (tid == 0) qm[row] = (r1[0] != 0.f) ? 1.f : 0.f;
  }
}

// ---------------- WMMA GEMM: (BSx512) x (512x512) + bias ----------------
// MODE 0: bf16 out                (Q, K projections)
// MODE 1: bf16 out, ReLU          (FFN up)
// MODE 2: f32 out + residual, row-masked by seq!=0   (FFN down)
// MODE 3: bf16 out transposed (out[n*BS+m])           (V projection)
//
// Block = 4 waves sharing one 64-column weight strip. The strip (16 kt rows x
// 4 KB, row stride 32 KB in the packed layout) is staged into LDS once per
// block with a single Tensor-Data-Mover 2D tile load (tensor_load_to_lds,
// waited with s_wait_tensorcnt), then each wave computes a 16x64 output tile
// for its own M-tile, reading B fragments from LDS.
template <int MODE>
__global__ void __launch_bounds__(128)
gemm512_kernel(const bf16_t* __restrict__ A, const bf16_t* __restrict__ Wp,
               const float* __restrict__ bias, void* __restrict__ outp,
               const float* __restrict__ residual, const int* __restrict__ seq) {
  __shared__ __align__(16) bf16_t lds_w[16 * 4 * 512];   // 64 KB weight strip
  int widx = threadIdx.x >> 5;
  int lane = threadIdx.x & 31;
  int ntg  = blockIdx.x & 7;                 // 64-column group
  int mt   = (blockIdx.x >> 3) * 4 + widx;   // this wave's 16-row tile

  const bf16_t* strip = Wp + (size_t)ntg * 4 * 512;   // 4 tiles/row, 32 rows/kt
#if __has_builtin(__builtin_amdgcn_tensor_load_to_lds) && \
    __has_builtin(__builtin_amdgcn_s_wait_tensorcnt)
  if (widx == 0) {
    unsigned long long ga = (unsigned long long)(uintptr_t)strip;
    unsigned lds_off = (unsigned)(uintptr_t)&lds_w[0];  // flat LDS addr[31:0]
    // D# group 0: count=1 | lds_addr | global_addr | type=2 (ISA 8.3)
    ui32x4 g0 = { 1u, lds_off, (unsigned)ga,
                  (unsigned)(ga >> 32) | (2u << 30) };
    // D# group 1 (ISA 8.4): data_size=3 (8B elems); tensor_dim0=4096 elems
    // (32 KB row span), tensor_dim1=16; tile_dim0=512 elems (4 KB),
    // tile_dim1=16; tensor_dim0_stride=4096 elems (32 KB).
    i32x8 g1 = { (int)(3u << 16),
                 (int)(4096u << 16),
                 (int)(16u << 16),
                 (int)(512u << 16),
                 16,
                 4096, 0, 0 };
    i32x4 gz  = { 0, 0, 0, 0 };               // 2-D tile: groups 2/3 unused
    i32x8 gz8 = { 0, 0, 0, 0, 0, 0, 0, 0 };   // trailing descriptor words
    __builtin_amdgcn_tensor_load_to_lds(g0, g1, gz, gz, gz8, 0);
    __builtin_amdgcn_s_wait_tensorcnt((short)0);
  }
#else
  // Portable fallback: cooperative copy of the 64 KB strip (16 rows x 4 KB).
  for (int i = threadIdx.x; i < 16 * 256; i += 128) {  // 4096 x 16B chunks
    int r = i >> 8, c = i & 255;
    *(uint4*)&lds_w[(size_t)r * 2048 + c * 8] =
        *(const uint4*)(strip + (size_t)r * 16384 + c * 8);
  }
#endif
  __syncthreads();

  v8f acc[4] = {v8f{}, v8f{}, v8f{}, v8f{}};
  const bf16_t* arow = A + (size_t)mt * 16 * 512;
  for (int kt = 0; kt < 16; ++kt) {
    v16bf a = load_a_frag(arow + kt * 32, 512);
    if (kt + 1 < 16) __builtin_prefetch(arow + (kt + 1) * 32, 0, 1);
#pragma unroll
    for (int j = 0; j < 4; ++j) {
      v16bf b = load_packed_b(&lds_w[(size_t)(kt * 4 + j) * 512]);
      acc[j] = wmma_bf16(a, b, acc[j]);
    }
  }

  int voff = ((lane >> 4) & 1) * 8;
#pragma unroll
  for (int j = 0; j < 4; ++j) {
    int n = (ntg * 4 + j) * 16 + (lane & 15);
    float bv = bias[n];
    if (MODE == 3) {
      v8bf t;
#pragma unroll
      for (int v = 0; v < 8; ++v) t[v] = (bf16_t)(acc[j][v] + bv);
      *(v8bf*)((bf16_t*)outp + (size_t)n * BS + (size_t)mt * 16 + voff) = t;
    } else {
#pragma unroll
      for (int v = 0; v < 8; ++v) {
        size_t m = (size_t)mt * 16 + voff + v;
        float val = acc[j][v] + bv;
        if (MODE == 1) val = fmaxf(val, 0.f);
        if (MODE == 2) {
          val += residual[m * 512 + n];
          val = (seq[m] != 0) ? val : 0.f;
          ((float*)outp)[m * 512 + n] = val;
        } else {
          ((bf16_t*)outp)[m * 512 + n] = (bf16_t)val;
        }
      }
    }
  }
}

// ---------------- fused attention ----------------
// One wave per (h, b, 16-query tile). Causal: only key tiles nt <= qt.
// scores (WMMA) -> LDS f32 -> softmax(+km/causal/qm masks) -> bf16 P in LDS
// -> O = P*V (WMMA, Vt row-major in K) -> out = O + xn (residual).
__global__ void __launch_bounds__(32)
attn_kernel(const bf16_t* __restrict__ Q, const bf16_t* __restrict__ Kb,
            const bf16_t* __restrict__ Vt, const float* __restrict__ km,
            const float* __restrict__ qm, const float* __restrict__ xn,
            float* __restrict__ outp) {
  __shared__ __align__(16) float  sc[16][512];
  __shared__ __align__(16) bf16_t pb[16][512];
  int blk  = blockIdx.x;
  int qt   = blk & 31;
  int b    = (blk >> 5) & 63;
  int h    = blk >> 11;
  int lane = threadIdx.x;
  int voff = ((lane >> 4) & 1) * 8;

  // zero P buffer (covers the tail half-tile of the last 32-wide K chunk)
  for (int i = lane; i < 16 * 512 / 2; i += 32) ((unsigned int*)pb)[i] = 0u;

  size_t qkbase = ((size_t)b * S_) * E_ + (size_t)h * D_;
  const bf16_t* Qp = Q + qkbase + (size_t)qt * 16 * E_;
  v16bf a0 = load_a_frag(Qp, E_);        // d = 0..31
  v16bf a1 = load_a_frag(Qp + 32, E_);   // d = 32..63
  int kl = (qt + 1) * 16;                // valid key length

  for (int nt = 0; nt <= qt; ++nt) {
    const bf16_t* Kp = Kb + qkbase + (size_t)nt * 16 * E_;
    v16bf b0 = load_bt_frag(Kp, E_);
    v16bf b1 = load_bt_frag(Kp + 32, E_);
    v8f s = {};
    s = wmma_bf16(a0, b0, s);
    s = wmma_bf16(a1, b1, s);
    int key = nt * 16 + (lane & 15);
    float kmv = km[(size_t)b * S_ + key];
#pragma unroll
    for (int v = 0; v < 8; ++v) {
      int q = qt * 16 + voff + v;
      float val = s[v] * 0.125f;         // 1/sqrt(D)
      if (kmv == 0.f) val = PADV;        // key padding mask
      if (key > q)    val = PADV;        // causal mask (diagonal tile)
      sc[voff + v][key] = val;
    }
  }
  __syncthreads();

  if (lane < 16) {
    int r = lane;
    int q = qt * 16 + r;
    float mx = -3.0e38f;
    for (int c = 0; c < kl; ++c) mx = fmaxf(mx, sc[r][c]);
    float sum = 0.f;
    for (int c = 0; c < kl; ++c) {
      float e = __expf(sc[r][c] - mx);
      sc[r][c] = e;
      sum += e;
    }
    float inv = qm[(size_t)b * S_ + q] / sum;   // fold query mask into scale
    for (int c = 0; c < kl; ++c) pb[r][c] = (bf16_t)(sc[r][c] * inv);
  }
  __syncthreads();

  int ktmax = (kl + 31) >> 5;
#pragma unroll
  for (int dt = 0; dt < 4; ++dt) {
    v8f o = {};
    for (int kt = 0; kt < ktmax; ++kt) {
      const bf16_t* p = &pb[lane & 15][kt * 32 + 2 * voff];
      v16bf a = join8(*(const v8bf*)p, *(const v8bf*)(p + 8));
      const bf16_t* vp = Vt + (size_t)(h * D_ + dt * 16) * BS + (size_t)b * S_ + kt * 32;
      v16bf bb = load_bt_frag(vp, BS);
      o = wmma_bf16(a, bb, o);
    }
    int d = h * D_ + dt * 16 + (lane & 15);
#pragma unroll
    for (int v = 0; v < 8; ++v) {
      size_t idx = ((size_t)b * S_ + qt * 16 + voff + v) * E_ + d;
      outp[idx] = o[v] + xn[idx];   // residual with (normalized) queries
    }
  }
}

// ---------------- host orchestration ----------------
extern "C" void kernel_launch(void* const* d_in, const int* in_sizes, int n_in,
                              void* d_out, int out_size, void* d_ws, size_t ws_size,
                              hipStream_t stream) {
  (void)in_sizes; (void)n_in; (void)out_size; (void)ws_size;
  const int*   seq      = (const int*)d_in[0];
  const float* item_emb = (const float*)d_in[1];
  const float* pos_emb  = (const float*)d_in[2];
  const float* ln_gamma = (const float*)d_in[3];
  const float* ln_beta  = (const float*)d_in[4];
  const float* wq = (const float*)d_in[5];
  const float* bq = (const float*)d_in[6];
  const float* wk = (const float*)d_in[7];
  const float* bk = (const float*)d_in[8];
  const float* wv = (const float*)d_in[9];
  const float* bv = (const float*)d_in[10];
  const float* w1 = (const float*)d_in[11];
  const float* b1 = (const float*)d_in[12];
  const float* w2 = (const float*)d_in[13];
  const float* b2 = (const float*)d_in[14];
  const float* fg = (const float*)d_in[15];
  const float* fb = (const float*)d_in[16];

  char* ws = (char*)d_ws;
  size_t off = 0;
  auto alloc = [&](size_t bytes) -> void* {
    void* p = ws + off;
    off += (bytes + 255) & ~(size_t)255;
    return p;
  };
  float*  x    = (float*)alloc((size_t)BS * E_ * 4);
  float*  xn   = (float*)alloc((size_t)BS * E_ * 4);
  float*  attn = (float*)alloc((size_t)BS * E_ * 4);
  bf16_t* xb   = (bf16_t*)alloc((size_t)BS * E_ * 2);
  bf16_t* xnb  = (bf16_t*)alloc((size_t)BS * E_ * 2);
  bf16_t* qbuf = (bf16_t*)alloc((size_t)BS * E_ * 2);
  bf16_t* kbuf = (bf16_t*)alloc((size_t)BS * E_ * 2);
  bf16_t* vt   = (bf16_t*)alloc((size_t)BS * E_ * 2);
  float*  km   = (float*)alloc((size_t)BS * 4);
  float*  qm   = (float*)alloc((size_t)BS * 4);
  bf16_t* pw   = (bf16_t*)alloc((size_t)L_ * 5 * 512 * 512 * 2);
  bf16_t* attnb = qbuf;   // Q dead after attention -> reuse
  bf16_t* hbuf  = kbuf;   // K dead after attention -> reuse

  // Pack all 10 weight matrices into WMMA-friendly bf16 layout.
  for (int l = 0; l < L_; ++l) {
    const float* srcs[5] = { wq + (size_t)l * E_ * E_, wk + (size_t)l * E_ * E_,
                             wv + (size_t)l * E_ * E_, w1 + (size_t)l * E_ * F_,
                             w2 + (size_t)l * F_ * E_ };
    for (int i = 0; i < 5; ++i)
      pack_w_kernel<<<1024, 256, 0, stream>>>(srcs[i],
                                              pw + ((size_t)l * 5 + i) * 512 * 512);
  }

  embed_kernel<<<(BS * E_) / 256, 256, 0, stream>>>(seq, item_emb, pos_emb, x);

  for (int l = 0; l < L_; ++l) {
    const bf16_t* pwl = pw + (size_t)l * 5 * 512 * 512;
    rowconv_kernel<<<BS, 256, 0, stream>>>(x, xb, km);
    ln_kernel<<<BS, 256, 0, stream>>>(x, ln_gamma + (size_t)l * S_ * E_,
                                      ln_beta + (size_t)l * S_ * E_, xn, xnb, qm);
    gemm512_kernel<0><<<4096, 128, 0, stream>>>(xnb, pwl + 0 * 262144, bq + l * E_,
                                                qbuf, nullptr, nullptr);
    gemm512_kernel<0><<<4096, 128, 0, stream>>>(xb,  pwl + 1 * 262144, bk + l * E_,
                                                kbuf, nullptr, nullptr);
    gemm512_kernel<3><<<4096, 128, 0, stream>>>(xb,  pwl + 2 * 262144, bv + l * E_,
                                                vt, nullptr, nullptr);
    attn_kernel<<<H_ * B_ * (S_ / 16), 32, 0, stream>>>(qbuf, kbuf, vt, km, qm, xn,
                                                        attn);
    rowconv_kernel<<<BS, 256, 0, stream>>>(attn, attnb, nullptr);
    gemm512_kernel<1><<<4096, 128, 0, stream>>>(attnb, pwl + 3 * 262144, b1 + l * F_,
                                                hbuf, nullptr, nullptr);
    gemm512_kernel<2><<<4096, 128, 0, stream>>>(hbuf, pwl + 4 * 262144, b2 + l * E_,
                                                x, attn, seq);
  }

  ln_kernel<<<BS, 256, 0, stream>>>(x, fg, fb, (float*)d_out, nullptr, nullptr);
}
